// GPT_90022514524335
// MI455X (gfx1250) — matrix operand order repaired and checked by reading, not compiled, not run on hardware
//
#include <hip/hip_runtime.h>
#include <math.h>

// ---------------------------------------------------------------------------
// CDNA5 (gfx1250) wave32 WMMA types & helpers
// ---------------------------------------------------------------------------
typedef __bf16 v16bf __attribute__((ext_vector_type(16)));
typedef float  v8f   __attribute__((ext_vector_type(8)));

union FragB {
  v16bf v;
  unsigned u[8];
  unsigned short s[16];
};

// Hardware bf16 conversion (v_cvt_bf16_f32 / v_cvt_pk_bf16_f32)
__device__ __forceinline__ unsigned short f2bf(float f) {
  union { __bf16 h; unsigned short s; } c;
  c.h = (__bf16)f;
  return c.s;
}
__device__ __forceinline__ unsigned pack2(float a, float b) {
  union { __bf16 h[2]; unsigned u; } c;
  c.h[0] = (__bf16)a;   // low  half = even K
  c.h[1] = (__bf16)b;   // high half = odd K
  return c.u;
}
__device__ __forceinline__ v8f wmma_bf16(const FragB& a, const FragB& b, v8f c) {
  // D = A(16x32 bf16) * B(32x16 bf16) + C(16x16 f32)
  return __builtin_amdgcn_wmma_f32_16x16x32_bf16(
      false, a.v, false, b.v, (short)0, c, false, false);
}

// ---------------------------------------------------------------------------
// Model constants (B=8, T=512 incl. cond token, D=768, H=12, dh=64, L=8, V=512)
// ---------------------------------------------------------------------------
#define MROWS 4096          // B*T
#define DMODEL 768
#define DMLP  3072
#define NVOC  512
#define TSEQ  512

// ---------------------------------------------------------------------------
// Embedding
// ---------------------------------------------------------------------------
__global__ void embed_kernel(const int* __restrict__ idx, const int* __restrict__ cond,
                             const float* __restrict__ tok, const float* __restrict__ ce,
                             const float* __restrict__ te, const float* __restrict__ pe,
                             float* __restrict__ x) {
  int i = blockIdx.x * blockDim.x + threadIdx.x;
  if (i >= 8 * TSEQ * DMODEL) return;
  int d = i % DMODEL;
  int t = (i / DMODEL) % TSEQ;
  int b = i / (DMODEL * TSEQ);
  float e, ty;
  if (t == 0) { e = ce[cond[b] * DMODEL + d]; ty = te[d]; }
  else        { e = tok[idx[b * 511 + t - 1] * DMODEL + d]; ty = te[DMODEL + d]; }
  x[i] = e + pe[t * DMODEL + d] + ty;
}

// ---------------------------------------------------------------------------
// LayerNorm: one block per row of 768, 3 elements per thread
// ---------------------------------------------------------------------------
__global__ __launch_bounds__(256)
void ln_kernel(const float* __restrict__ X, const float* __restrict__ g,
               const float* __restrict__ b, float* __restrict__ Y) {
  __shared__ float red[256];
  const int tid = threadIdx.x;
  const float* xr = X + (size_t)blockIdx.x * DMODEL;
  float v0 = xr[tid], v1 = xr[tid + 256], v2 = xr[tid + 512];
  red[tid] = v0 + v1 + v2;
  __syncthreads();
  for (int off = 128; off > 0; off >>= 1) {
    if (tid < off) red[tid] += red[tid + off];
    __syncthreads();
  }
  float mean = red[0] * (1.0f / 768.0f);
  __syncthreads();
  float d0 = v0 - mean, d1 = v1 - mean, d2 = v2 - mean;
  red[tid] = d0 * d0 + d1 * d1 + d2 * d2;
  __syncthreads();
  for (int off = 128; off > 0; off >>= 1) {
    if (tid < off) red[tid] += red[tid + off];
    __syncthreads();
  }
  float rstd = rsqrtf(red[0] * (1.0f / 768.0f) + 1e-5f);
  float* yr = Y + (size_t)blockIdx.x * DMODEL;
  yr[tid]       = d0 * rstd * g[tid]       + b[tid];
  yr[tid + 256] = d1 * rstd * g[tid + 256] + b[tid + 256];
  yr[tid + 512] = d2 * rstd * g[tid + 512] + b[tid + 512];
}

// ---------------------------------------------------------------------------
// GEMM: C[M,N] = act(A[M,K] @ W[K,N] + bias) (+ res). fp32 in/out, bf16 WMMA.
// Block tile 64x128x64, 8 waves (2x4), 32x32 per wave; 8 WMMAs/wave per tile.
// ---------------------------------------------------------------------------
#define GT_M 64
#define GT_N 128
#define GT_K 64
#define LDSS  66   // shorts per LDS row (64 + 2 pad)
#define LDSSU 33   // dwords per LDS row

__global__ __launch_bounds__(256)
void gemm_kernel(const float* __restrict__ A, const float* __restrict__ W,
                 const float* bias, const float* res, float* C,
                 int M, int N, int K, int act) {
  __shared__ __align__(16) unsigned short Asm[GT_M * LDSS];
  __shared__ __align__(16) unsigned short Bsm[GT_N * LDSS];

  const int tid  = threadIdx.x;
  const int lane = tid & 31;
  const int wid  = tid >> 5;
  const int wm   = wid & 1;     // wave row (0..1)  -> 32 rows each
  const int wn   = wid >> 1;    // wave col (0..3)  -> 32 cols each
  const int m0   = blockIdx.y * GT_M;
  const int n0   = blockIdx.x * GT_N;
  const int ln15 = lane & 15;
  const int g    = lane >> 4;

  v8f acc[2][2] = {};

  const int aRow = tid >> 2;          // 0..63
  const int aSeg = (tid & 3) * 16;    // 0,16,32,48
  const int bN   = (tid & 31) * 4;    // 0..124
  const int bKb  = (tid >> 5) * 8;    // 0..56

  for (int k0 = 0; k0 < K; k0 += GT_K) {
    // --- stage A tile (fp32 -> bf16), row-major [64][64], packed dwords ---
    {
      const float* src = A + (size_t)(m0 + aRow) * K + k0 + aSeg;
      unsigned* dst = (unsigned*)&Asm[aRow * LDSS + aSeg];
#pragma unroll
      for (int i = 0; i < 8; ++i) dst[i] = pack2(src[2 * i], src[2 * i + 1]);
    }
    // --- stage W tile transposed: Bsm[n][k] = W[k0+k][n0+n], pair-packed ---
#pragma unroll
    for (int r = 0; r < 4; ++r) {
      int kk = bKb + r * 2;
      const float4 w0 = *(const float4*)(W + (size_t)(k0 + kk) * N + n0 + bN);
      const float4 w1 = *(const float4*)(W + (size_t)(k0 + kk + 1) * N + n0 + bN);
      unsigned* bu = (unsigned*)Bsm;
      int kd = kk >> 1;
      bu[(bN + 0) * LDSSU + kd] = pack2(w0.x, w1.x);
      bu[(bN + 1) * LDSSU + kd] = pack2(w0.y, w1.y);
      bu[(bN + 2) * LDSSU + kd] = pack2(w0.z, w1.z);
      bu[(bN + 3) * LDSSU + kd] = pack2(w0.w, w1.w);
    }
    if (k0 + GT_K < K) {
      __builtin_prefetch(A + (size_t)(m0 + aRow) * K + k0 + GT_K + aSeg, 0, 0);
      __builtin_prefetch(W + (size_t)(k0 + GT_K + bKb) * N + n0 + bN, 0, 0);
    }
    __syncthreads();

    // --- two K=32 substeps per LDS tile: 8 WMMAs per wave between barriers ---
#pragma unroll
    for (int ks2 = 0; ks2 < 2; ++ks2) {
      FragB af[2], bf[2];
#pragma unroll
      for (int i = 0; i < 2; ++i) {
        const unsigned* ap =
            (const unsigned*)&Asm[(wm * 32 + i * 16 + ln15) * LDSS];
#pragma unroll
        for (int j = 0; j < 4; ++j) {
          af[i].u[j]     = ap[ks2 * 16 + 8 * g + j];      // K = ks2*32 + 16g+2j
          af[i].u[4 + j] = ap[ks2 * 16 + 8 * g + 4 + j];  // K = ks2*32 + 16g+8+2j
        }
      }
#pragma unroll
      for (int j2 = 0; j2 < 2; ++j2) {
        const unsigned* bp =
            (const unsigned*)&Bsm[(wn * 32 + j2 * 16 + ln15) * LDSS];
#pragma unroll
        for (int j = 0; j < 8; ++j)
          bf[j2].u[j] = bp[ks2 * 16 + 8 * g + j];         // K = ks2*32 + 16g+t
      }
#pragma unroll
      for (int i = 0; i < 2; ++i)
#pragma unroll
        for (int j = 0; j < 2; ++j)
          acc[i][j] = wmma_bf16(af[i], bf[j], acc[i][j]);
    }
    __syncthreads();
  }

  // --- epilogue: C layout lane = N%16, VGPR r -> row r + 8*(lane>=16) ---
#pragma unroll
  for (int i = 0; i < 2; ++i) {
    int rowb = m0 + wm * 32 + i * 16 + 8 * g;
#pragma unroll
    for (int j = 0; j < 2; ++j) {
      int col = n0 + wn * 32 + j * 16 + ln15;
      float bb = bias ? bias[col] : 0.0f;
#pragma unroll
      for (int r = 0; r < 8; ++r) {
        float vl = acc[i][j][r] + bb;
        if (act == 1) vl = 0.5f * vl * (1.0f + erff(vl * 0.70710678118654752f));
        size_t off = (size_t)(rowb + r) * N + col;
        if (res) vl += res[off];
        C[off] = vl;
      }
    }
  }
}

// ---------------------------------------------------------------------------
// Fused attention: one wave per (batch, head, 16-query tile).
//   S = mask(QK^T/8) -> softmax -> Y = P V, all WMMA, T=512, dh=64.
// ---------------------------------------------------------------------------
#define SROW 513          // fp32 score row stride (pad)
#define PROW 520          // bf16 P row stride in shorts (260 dwords)
#define VTS  34           // bf16 Vt row stride in shorts (17 dwords)

__global__ __launch_bounds__(32)
void attn_kernel(const float* __restrict__ Q, const float* __restrict__ Km,
                 const float* __restrict__ V, float* __restrict__ Y) {
  __shared__ __align__(16) float          S[16 * SROW];
  __shared__ __align__(16) unsigned short P[16 * PROW];
  __shared__ __align__(16) unsigned short Vt[64 * VTS];

  const int lane = threadIdx.x;
  const int ln15 = lane & 15;
  const int g    = lane >> 4;

  const int blk = blockIdx.x;
  const int qt  = blk & 31;            // 32 query tiles
  const int rst = blk >> 5;
  const int hh  = rst % 12;
  const int bb  = rst / 12;
  const int q0  = qt * 16;

  const size_t base = (size_t)bb * TSEQ * DMODEL + hh * 64;
  const float* qp = Q + base;
  const float* kp = Km + base;
  const float* vp = V + base;
  float* yp = Y + base;

  // ---- Q fragments (16x64 = two 16x32 K-slices), loaded once ----
  FragB aQ[2];
#pragma unroll
  for (int s = 0; s < 2; ++s) {
    const float* qr = qp + (size_t)(q0 + ln15) * DMODEL + s * 32;
#pragma unroll
    for (int j = 0; j < 4; ++j) {
      aQ[s].u[j]     = pack2(qr[16 * g + 2 * j],     qr[16 * g + 2 * j + 1]);
      aQ[s].u[4 + j] = pack2(qr[16 * g + 8 + 2 * j], qr[16 * g + 8 + 2 * j + 1]);
    }
  }

  // ---- scores: S[16,512] = mask(Q K^T / sqrt(64)) ----
  for (int kt = 0; kt < 32; ++kt) {
    v8f sc = {};
#pragma unroll
    for (int s = 0; s < 2; ++s) {
      FragB bK;  // column n of K^T == row (kt*16+n) of K: contiguous dh slice
      const float* kr = kp + (size_t)(kt * 16 + ln15) * DMODEL + s * 32 + 16 * g;
#pragma unroll
      for (int jj = 0; jj < 4; ++jj) {
        float4 kv = *(const float4*)(kr + 4 * jj);
        bK.u[2 * jj]     = pack2(kv.x, kv.y);
        bK.u[2 * jj + 1] = pack2(kv.z, kv.w);
      }
      sc = wmma_bf16(aQ[s], bK, sc);
    }
    int col = kt * 16 + ln15;
#pragma unroll
    for (int r = 0; r < 8; ++r) {
      int row = r + 8 * g;
      float val = sc[r] * 0.125f;
      if (col > q0 + row) val = -1e30f;   // causal mask
      S[row * SROW + col] = val;
    }
  }
  __syncthreads();

  // ---- softmax: lanes (r, r+16) split each row's 512 cols ----
  {
    const int row = ln15;
    const int c0 = g * 256;
    float mx = -1e30f;
    for (int c = c0; c < c0 + 256; ++c) mx = fmaxf(mx, S[row * SROW + c]);
    mx = fmaxf(mx, __shfl_xor(mx, 16, 32));
    float sum = 0.0f;
    for (int c = c0; c < c0 + 256; ++c) {
      float e = expf(S[row * SROW + c] - mx);
      S[row * SROW + c] = e;
      sum += e;
    }
    sum += __shfl_xor(sum, 16, 32);
    float inv = 1.0f / sum;
    for (int c = c0; c < c0 + 256; c += 2)
      *(unsigned*)&P[row * PROW + c] =
          pack2(S[row * SROW + c] * inv, S[row * SROW + c + 1] * inv);
  }
  __syncthreads();

  // ---- Y = P(16x512) @ V(512x64): 16 K-steps of 32, 4 output tiles ----
  v8f acc[4] = {};
  for (int ks = 0; ks < 16; ++ks) {
    __syncthreads();  // previous Vt reads done
    // stage Vt[n][kk] = V[ks*32+kk][n] (bf16), pair-packed dword stores,
    // coalesced global reads (consecutive lanes -> consecutive n)
    unsigned* vtu = (unsigned*)Vt;
    for (int e = 0; e < 32; ++e) {
      int id = e * 32 + lane;        // 0..1023
      int n = id & 63, k2 = id >> 6; // k2 = K pair index 0..15
      float a = vp[(size_t)(ks * 32 + 2 * k2) * DMODEL + n];
      float b = vp[(size_t)(ks * 32 + 2 * k2 + 1) * DMODEL + n];
      vtu[n * 17 + k2] = pack2(a, b);
    }
    __syncthreads();

    FragB aP;
    const unsigned* pr = (const unsigned*)&P[ln15 * PROW];
#pragma unroll
    for (int j = 0; j < 4; ++j) {
      aP.u[j]     = pr[ks * 16 + 8 * g + j];
      aP.u[4 + j] = pr[ks * 16 + 8 * g + 4 + j];
    }
#pragma unroll
    for (int jt = 0; jt < 4; ++jt) {
      FragB bV;
      const unsigned* vr = (const unsigned*)&Vt[(jt * 16 + ln15) * VTS];
#pragma unroll
      for (int j = 0; j < 8; ++j) bV.u[j] = vr[8 * g + j];
      acc[jt] = wmma_bf16(aP, bV, acc[jt]);
    }
  }

  // ---- write Y tile (16 x 64) ----
#pragma unroll
  for (int jt = 0; jt < 4; ++jt)
#pragma unroll
    for (int r = 0; r < 8; ++r)
      yp[(size_t)(q0 + r + 8 * g) * DMODEL + jt * 16 + ln15] = acc[jt][r];
}

// ---------------------------------------------------------------------------
// Strip conditioning position: out[b, t, v] = logits[b, t+1, v]
// ---------------------------------------------------------------------------
__global__ void slice_kernel(const float* __restrict__ logits, float* __restrict__ out) {
  int i = blockIdx.x * blockDim.x + threadIdx.x;
  if (i >= 8 * 511 * NVOC) return;
  int v = i % NVOC;
  int t = (i / NVOC) % 511;
  int b = i / (NVOC * 511);
  out[i] = logits[((size_t)(b * TSEQ + t + 1)) * NVOC + v];
}

// ---------------------------------------------------------------------------
// Host driver
// ---------------------------------------------------------------------------
extern "C" void kernel_launch(void* const* d_in, const int* in_sizes, int n_in,
                              void* d_out, int out_size, void* d_ws, size_t ws_size,
                              hipStream_t stream) {
  (void)in_sizes; (void)n_in; (void)out_size; (void)ws_size;

  const int*   idx      = (const int*)d_in[0];
  const int*   cond     = (const int*)d_in[1];
  const float* tok_emb  = (const float*)d_in[2];
  const float* cond_emb = (const float*)d_in[3];
  const float* type_emb = (const float*)d_in[4];
  const float* pos_emb  = (const float*)d_in[5];
  const float* ln1_g    = (const float*)d_in[6];
  const float* ln1_b    = (const float*)d_in[7];
  const float* Wq       = (const float*)d_in[8];
  const float* bq       = (const float*)d_in[9];
  const float* Wk       = (const float*)d_in[10];
  const float* bk       = (const float*)d_in[11];
  const float* Wv       = (const float*)d_in[12];
  const float* bv       = (const float*)d_in[13];
  const float* Wp       = (const float*)d_in[14];
  const float* bp       = (const float*)d_in[15];
  const float* ln2_g    = (const float*)d_in[16];
  const float* ln2_b    = (const float*)d_in[17];
  const float* W1       = (const float*)d_in[18];
  const float* b1       = (const float*)d_in[19];
  const float* W2       = (const float*)d_in[20];
  const float* b2       = (const float*)d_in[21];
  const float* lnf_g    = (const float*)d_in[22];
  const float* lnf_b    = (const float*)d_in[23];
  const float* head_w   = (const float*)d_in[24];

  const int M = MROWS, D = DMODEL, D4 = DMLP, Vv = NVOC, L = 8;
  float* x      = (float*)d_ws;
  float* h      = x + (size_t)M * D;
  float* q      = h + (size_t)M * D;
  float* k      = q + (size_t)M * D;
  float* v      = k + (size_t)M * D;
  float* y      = v + (size_t)M * D;
  float* t4     = y + (size_t)M * D;
  float* logits = t4 + (size_t)M * D4;

  embed_kernel<<<(M * D + 255) / 256, 256, 0, stream>>>(
      idx, cond, tok_emb, cond_emb, type_emb, pos_emb, x);

  dim3 gD(D / GT_N, M / GT_M);    // (6, 64)
  dim3 g4(D4 / GT_N, M / GT_M);   // (24, 64)
  dim3 gV(Vv / GT_N, M / GT_M);   // (4, 64)

  for (int l = 0; l < L; ++l) {
    size_t wo = (size_t)l * D * D;
    ln_kernel<<<M, 256, 0, stream>>>(x, ln1_g + l * D, ln1_b + l * D, h);
    gemm_kernel<<<gD, 256, 0, stream>>>(h, Wq + wo, bq + l * D, nullptr, q, M, D, D, 0);
    gemm_kernel<<<gD, 256, 0, stream>>>(h, Wk + wo, bk + l * D, nullptr, k, M, D, D, 0);
    gemm_kernel<<<gD, 256, 0, stream>>>(h, Wv + wo, bv + l * D, nullptr, v, M, D, D, 0);
    attn_kernel<<<8 * 12 * 32, 32, 0, stream>>>(q, k, v, y);
    // residual base is post-ln h (reference quirk): x = h + y@Wp + bp
    gemm_kernel<<<gD, 256, 0, stream>>>(y, Wp + wo, bp + l * D, h, x, M, D, D, 0);
    ln_kernel<<<M, 256, 0, stream>>>(x, ln2_g + l * D, ln2_b + l * D, h);
    gemm_kernel<<<g4, 256, 0, stream>>>(h, W1 + (size_t)l * D * D4, b1 + l * D4,
                                        nullptr, t4, M, D4, D, 1 /*gelu*/);
    gemm_kernel<<<gD, 256, 0, stream>>>(t4, W2 + (size_t)l * D4 * D, b2 + l * D,
                                        x, x, M, D, D4, 0);
  }

  ln_kernel<<<M, 256, 0, stream>>>(x, lnf_g, lnf_b, h);
  gemm_kernel<<<gV, 256, 0, stream>>>(h, head_w, nullptr, nullptr, logits, M, Vv, D, 0);
  slice_kernel<<<(8 * 511 * NVOC + 255) / 256, 256, 0, stream>>>(logits, (float*)d_out);
}
